// MHA_10101763080199
// MI455X (gfx1250) — matrix operand order repaired and checked by reading, not compiled, not run on hardware
//
#include <hip/hip_runtime.h>

#define NB 4
#define NS 1024
#define ND 1024
#define NH 16
#define NDH 64

typedef __attribute__((ext_vector_type(16))) __bf16 v16bf;
typedef __attribute__((ext_vector_type(8)))  float  v8f;

__device__ __forceinline__ unsigned short f2bf(float f) {
  union { float f; unsigned int u; } x; x.f = f;
  unsigned int r = x.u + 0x7FFFu + ((x.u >> 16) & 1u);
  return (unsigned short)(r >> 16);
}

// A-fragment (16x32 bf16): lane row = M, K = 2*kb+0..15 contiguous, kb = (lane<16?0:8)
__device__ __forceinline__ v16bf load_frag_a(const unsigned short* __restrict__ rp,
                                             int kstart, int kb) {
  union { uint4 u[2]; v16bf v; } r;
  r.u[0] = *reinterpret_cast<const uint4*>(rp + kstart + 2 * kb);
  r.u[1] = *reinterpret_cast<const uint4*>(rp + kstart + 2 * kb + 8);
  return r.v;
}
// B-fragment (32x16 bf16): lane col = N, K = kb2+0..15 contiguous, kb2 = (lane<16?0:16)
__device__ __forceinline__ v16bf load_frag_b(const unsigned short* __restrict__ rp,
                                             int kstart, int kb2) {
  union { uint4 u[2]; v16bf v; } r;
  const uint4* p = reinterpret_cast<const uint4*>(rp + kstart + kb2);
  r.u[0] = p[0]; r.u[1] = p[1];
  return r.v;
}
__device__ __forceinline__ v8f wmma_bf16(v16bf a, v16bf b, v8f c) {
  return __builtin_amdgcn_wmma_f32_16x16x32_bf16(false, a, false, b, (short)0, c,
                                                 false, false);
}

__global__ void f32_to_bf16_kernel(const float* __restrict__ src,
                                   unsigned short* __restrict__ dst, int n) {
  int i = blockIdx.x * blockDim.x + threadIdx.x;
  if (i < n) dst[i] = f2bf(src[i]);
}

// qkv = x @ Wqkv^T + bqkv, fused RoPE on q,k; writes Q[b,h,s,dh], K[b,h,s,dh], VT[b,h,dh,s]
// Wave tile: 32 (M) x 64 (N). Block = 8 waves -> 256 rows x 64 cols.
__global__ __launch_bounds__(256) void qkv_rope_kernel(
    const unsigned short* __restrict__ Xb, const unsigned short* __restrict__ Wb,
    const float* __restrict__ bias,
    unsigned short* __restrict__ Qo, unsigned short* __restrict__ Ko,
    unsigned short* __restrict__ VTo) {
  const int lane = threadIdx.x & 31;
  const int wave = threadIdx.x >> 5;
  const int kb  = (lane < 16) ? 0 : 8;
  const int kb2 = (lane < 16) ? 0 : 16;
  const int mo  = (lane < 16) ? 0 : 8;
  const int nsel = lane & 15;
  const int row0 = blockIdx.y * 256 + wave * 32;      // token-row tile (M)
  const int col0 = blockIdx.x * 64;                   // e tile (N), head-aligned
  const unsigned short* arow0 = Xb + (size_t)(row0 + nsel) * ND;
  const unsigned short* arow1 = Xb + (size_t)(row0 + 16 + nsel) * ND;
  const unsigned short* brow  = Wb + (size_t)(col0 + nsel) * ND;

  v8f acc[2][4] = {{v8f{}, v8f{}, v8f{}, v8f{}}, {v8f{}, v8f{}, v8f{}, v8f{}}};
  for (int k = 0; k < ND; k += 32) {
    __builtin_prefetch(arow0 + k + 128, 0, 3);
    __builtin_prefetch(arow1 + k + 128, 0, 3);
    v16bf a0 = load_frag_a(arow0, k, kb);
    v16bf a1 = load_frag_a(arow1, k, kb);
#pragma unroll
    for (int j = 0; j < 4; ++j) {
      v16bf b = load_frag_b(brow + (size_t)(j * 16) * ND, k, kb2);
      acc[0][j] = wmma_bf16(a0, b, acc[0][j]);
      acc[1][j] = wmma_bf16(a1, b, acc[1][j]);
    }
  }
#pragma unroll
  for (int j = 0; j < 4; ++j) {
    float bj = bias[col0 + j * 16 + nsel];
    acc[0][j] += bj;
    acc[1][j] += bj;
  }

  const int which = col0 >> 10;            // 0=q 1=k 2=v
  const int h = (col0 & 1023) >> 6;

#pragma unroll
  for (int rt = 0; rt < 2; ++rt) {
    const int rowt = row0 + rt * 16;
    if (which < 2) {
      // RoPE: pair (dh, dh+32) lives in acc[rt][j], acc[rt][j+2], same lane
#pragma unroll
      for (int j = 0; j < 2; ++j) {
        float fidx = (float)(j * 16 + nsel);
        float invf = __expf(-0.28782313662425572f * fidx);  // ln(1e4)/32
#pragma unroll
        for (int i = 0; i < 8; ++i) {
          int r = rowt + mo + i;
          float ang = (float)(r & (NS - 1)) * invf;
          float sn, cs; __sincosf(ang, &sn, &cs);
          float x1 = acc[rt][j][i], x2 = acc[rt][j + 2][i];
          acc[rt][j][i]     = x1 * cs - x2 * sn;
          acc[rt][j + 2][i] = x2 * cs + x1 * sn;
        }
      }
      unsigned short* dst = (which == 0) ? Qo : Ko;
#pragma unroll
      for (int j = 0; j < 4; ++j) {
        int dh = j * 16 + nsel;
#pragma unroll
        for (int i = 0; i < 8; ++i) {
          int r = rowt + mo + i;
          int b = r >> 10, srow = r & (NS - 1);
          dst[((size_t)((b * NH + h) * NS + srow)) * NDH + dh] = f2bf(acc[rt][j][i]);
        }
      }
    } else {
      // V transposed: lane holds fixed dh, 8 consecutive s -> one 16B store
      int rbase = rowt + mo;
      int b = rbase >> 10, srow = rbase & (NS - 1);
#pragma unroll
      for (int j = 0; j < 4; ++j) {
        int dh = j * 16 + nsel;
        union { unsigned short s[8]; uint4 u; } ob;
#pragma unroll
        for (int i = 0; i < 8; ++i) ob.s[i] = f2bf(acc[rt][j][i]);
        size_t idx = ((size_t)((b * NH + h) * NDH + dh)) * NS + srow;
        *reinterpret_cast<uint4*>(VTo + idx) = ob.u;
      }
    }
  }
}

// Flash attention: one wave per (b,h,16-query tile). S^T = K*Q^T, Out^T = V^T*P^T.
__global__ __launch_bounds__(128) void attn_kernel(
    const unsigned short* __restrict__ Qi, const unsigned short* __restrict__ Ki,
    const unsigned short* __restrict__ VTi, unsigned short* __restrict__ AO) {
  const int lane = threadIdx.x & 31;
  const int wave = threadIdx.x >> 5;
  const int gw = blockIdx.x * 4 + wave;     // 0..4095
  const int tTile = gw & 63;
  const int bh = gw >> 6;                   // 0..63
  const int b = bh >> 4, h = bh & 15;
  const int t0 = tTile * 16;
  const int kb  = (lane < 16) ? 0 : 8;
  const int kb2 = (lane < 16) ? 0 : 16;
  const int mo  = (lane < 16) ? 0 : 8;
  const int nsel = lane & 15;
  const int tq = t0 + nsel;                 // this lane's query column

  const unsigned short* qrow = Qi + ((size_t)bh * NS + tq) * NDH;
  v16bf qf0 = load_frag_b(qrow, 0, kb2);    // dh 0..31   (B operand of S^T)
  v16bf qf1 = load_frag_b(qrow, 32, kb2);   // dh 32..63
  const unsigned short* Kbase = Ki + (size_t)bh * NS * NDH;
  const unsigned short* Vbase = VTi + (size_t)bh * NDH * NS;

  float m_run = -3.0e38f, l_run = 0.0f;
  v8f ot[4] = {v8f{}, v8f{}, v8f{}, v8f{}};

  const int nchunks = (t0 + 47) >> 5;       // causal: only keys <= t0+15
  for (int c = 0; c < nchunks; ++c) {
    const int s0 = c * 32;
    v8f st[2];
#pragma unroll
    for (int kt = 0; kt < 2; ++kt) {
      const unsigned short* krow = Kbase + (size_t)(s0 + kt * 16 + nsel) * NDH;
      v8f z = {};
      z = wmma_bf16(load_frag_a(krow, 0, kb), qf0, z);
      z = wmma_bf16(load_frag_a(krow, 32, kb), qf1, z);
      st[kt] = z;
    }
    // scale + causal mask + online softmax (lane = query column)
    float p[16];
    float vmax = -3.0e38f;
#pragma unroll
    for (int kt = 0; kt < 2; ++kt)
#pragma unroll
      for (int i = 0; i < 8; ++i) {
        int sg = s0 + kt * 16 + mo + i;
        float v = st[kt][i] * 0.125f + ((sg > tq) ? -10000.0f : 0.0f);
        p[kt * 8 + i] = v;
        vmax = fmaxf(vmax, v);
      }
    vmax = fmaxf(vmax, __shfl_xor(vmax, 16, 32));
    float m_new = fmaxf(m_run, vmax);
    float corr = __expf(m_run - m_new);
    float lsum = 0.0f;
#pragma unroll
    for (int i = 0; i < 16; ++i) { p[i] = __expf(p[i] - m_new); lsum += p[i]; }
    l_run = l_run * corr + lsum + __shfl_xor(lsum, 16, 32);
    m_run = m_new;
#pragma unroll
    for (int j = 0; j < 4; ++j) ot[j] *= corr;

    // P^T as B-fragment: pack to bf16, swap halves across lane^16
    unsigned int pk0[4], pk1[4], ex0[4], ex1[4];
#pragma unroll
    for (int i = 0; i < 4; ++i) {
      pk0[i] = (unsigned)f2bf(p[2 * i]) | ((unsigned)f2bf(p[2 * i + 1]) << 16);
      pk1[i] = (unsigned)f2bf(p[8 + 2 * i]) | ((unsigned)f2bf(p[9 + 2 * i]) << 16);
    }
#pragma unroll
    for (int i = 0; i < 4; ++i) {
      ex0[i] = (unsigned)__shfl_xor((int)pk0[i], 16, 32);
      ex1[i] = (unsigned)__shfl_xor((int)pk1[i], 16, 32);
    }
    union { unsigned int u[8]; v16bf v; } pf;
    const bool lo = lane < 16;
#pragma unroll
    for (int i = 0; i < 4; ++i) { pf.u[i] = lo ? pk0[i] : ex1[i]; pf.u[4 + i] = lo ? ex0[i] : pk1[i]; }

    // Out^T += V^T x P^T
#pragma unroll
    for (int j = 0; j < 4; ++j) {
      const unsigned short* vrow = Vbase + (size_t)(j * 16 + nsel) * NS;
      ot[j] = wmma_bf16(load_frag_a(vrow, s0, kb), pf.v, ot[j]);
    }
  }

  float rl = 1.0f / l_run;
#pragma unroll
  for (int j = 0; j < 4; ++j) {
    union { unsigned short s[8]; uint4 u; } ob;
#pragma unroll
    for (int i = 0; i < 8; ++i) ob.s[i] = f2bf(ot[j][i] * rl);
    int d0 = h * 64 + j * 16 + mo;
    size_t idx = ((size_t)(b * NS + tq)) * ND + d0;
    *reinterpret_cast<uint4*>(AO + idx) = ob.u;
  }
}

// out = AO @ Wout^T + bout  (fp32 output). Wave tile: 32 (M) x 64 (N).
__global__ __launch_bounds__(256) void out_proj_kernel(
    const unsigned short* __restrict__ AO, const unsigned short* __restrict__ Wb,
    const float* __restrict__ bias, float* __restrict__ out) {
  const int lane = threadIdx.x & 31;
  const int wave = threadIdx.x >> 5;
  const int kb  = (lane < 16) ? 0 : 8;
  const int kb2 = (lane < 16) ? 0 : 16;
  const int mo  = (lane < 16) ? 0 : 8;
  const int nsel = lane & 15;
  const int row0 = blockIdx.y * 256 + wave * 32;
  const int col0 = blockIdx.x * 64;
  const unsigned short* arow0 = AO + (size_t)(row0 + nsel) * ND;
  const unsigned short* arow1 = AO + (size_t)(row0 + 16 + nsel) * ND;
  const unsigned short* brow  = Wb + (size_t)(col0 + nsel) * ND;

  v8f acc[2][4] = {{v8f{}, v8f{}, v8f{}, v8f{}}, {v8f{}, v8f{}, v8f{}, v8f{}}};
  for (int k = 0; k < ND; k += 32) {
    __builtin_prefetch(arow0 + k + 128, 0, 3);
    __builtin_prefetch(arow1 + k + 128, 0, 3);
    v16bf a0 = load_frag_a(arow0, k, kb);
    v16bf a1 = load_frag_a(arow1, k, kb);
#pragma unroll
    for (int j = 0; j < 4; ++j) {
      v16bf b = load_frag_b(brow + (size_t)(j * 16) * ND, k, kb2);
      acc[0][j] = wmma_bf16(a0, b, acc[0][j]);
      acc[1][j] = wmma_bf16(a1, b, acc[1][j]);
    }
  }
#pragma unroll
  for (int j = 0; j < 4; ++j) {
    int e = col0 + j * 16 + nsel;
    float bj = bias[e];
#pragma unroll
    for (int rt = 0; rt < 2; ++rt)
#pragma unroll
      for (int i = 0; i < 8; ++i) {
        out[(size_t)(row0 + rt * 16 + mo + i) * ND + e] = acc[rt][j][i] + bj;
      }
  }
}

extern "C" void kernel_launch(void* const* d_in, const int* in_sizes, int n_in,
                              void* d_out, int out_size, void* d_ws, size_t ws_size,
                              hipStream_t stream) {
  const float* x    = (const float*)d_in[0];   // [4,1024,1024]
  const float* Wqkv = (const float*)d_in[1];   // [3072,1024]
  const float* bqkv = (const float*)d_in[2];   // [3072]
  const float* Wout = (const float*)d_in[3];   // [1024,1024]
  const float* bout = (const float*)d_in[4];   // [1024]
  float* out = (float*)d_out;

  char* ws = (char*)d_ws;
  const size_t n_x = (size_t)NB * NS * ND;         // 4194304
  const size_t n_wqkv = (size_t)3 * ND * ND;       // 3145728
  const size_t n_wout = (size_t)ND * ND;           // 1048576
  unsigned short* xb    = (unsigned short*)(ws);
  unsigned short* wqkvb = (unsigned short*)(ws + 2 * n_x);
  unsigned short* woutb = (unsigned short*)(ws + 2 * (n_x + n_wqkv));
  unsigned short* Qp    = (unsigned short*)(ws + 2 * (n_x + n_wqkv + n_wout));
  unsigned short* Kp    = Qp + n_x;
  unsigned short* VTp   = Kp + n_x;
  unsigned short* AOp   = VTp + n_x;

  f32_to_bf16_kernel<<<(int)((n_x + 255) / 256), 256, 0, stream>>>(x, xb, (int)n_x);
  f32_to_bf16_kernel<<<(int)((n_wqkv + 255) / 256), 256, 0, stream>>>(Wqkv, wqkvb, (int)n_wqkv);
  f32_to_bf16_kernel<<<(int)((n_wout + 255) / 256), 256, 0, stream>>>(Wout, woutb, (int)n_wout);

  // M=4096 x N=3072: block = 8 waves x (32x64) = 256 rows x 64 cols
  qkv_rope_kernel<<<dim3(48, 16), 256, 0, stream>>>(xb, wqkvb, bqkv, Qp, Kp, VTp);

  // 4096 waves = (b,h) x 64 query tiles; 4 waves/block
  attn_kernel<<<dim3(1024), 128, 0, stream>>>(Qp, Kp, VTp, AOp);

  // M=4096 x N=1024: 16 x 16 blocks
  out_proj_kernel<<<dim3(16, 16), 256, 0, stream>>>(AOp, woutb, bout, out);
}
